// MSAColumnAttention_962072674749
// MI455X (gfx1250) — matrix operand order repaired and checked by reading, not compiled, not run on hardware
//
#include <hip/hip_runtime.h>
#include <hip/hip_bf16.h>

typedef __bf16 bf16_t;
typedef __attribute__((ext_vector_type(16))) __bf16 v16bf;
typedef __attribute__((ext_vector_type(8)))  __bf16 v8bf;
typedef __attribute__((ext_vector_type(8)))  float   v8f;

#define WMMA_BF16(a, b, c) \
  __builtin_amdgcn_wmma_f32_16x16x32_bf16(false, (a), false, (b), (short)0, (c), false, false)

// ---------------- sizes ----------------
#define S_DIM 256
#define I_DIM 256
#define C_S   256
#define NHEAD 8
#define HDIM  32
#define TOTAL 256          // NHEAD*HDIM
#define ROWS  65536        // S_DIM*I_DIM
#define QK_SCALE 0.17677669529663687f   // 1/sqrt(32)

// ---------------- workspace layout (bytes) ----------------
#define OFF_X    ((size_t)0)                          // bf16 [65536][256]      32 MB
#define OFF_WT   (OFF_X   + (size_t)ROWS*256*2)       // bf16 [1024][256]
#define OFF_WOT  (OFF_WT  + (size_t)1024*256*2)       // bf16 [256][256]
#define OFF_Q    (OFF_WOT + (size_t)256*256*2)        // bf16 [I][H][S][D]      32 MB
#define OFF_K    (OFF_Q   + (size_t)ROWS*256*2)
#define OFF_V    (OFF_K   + (size_t)ROWS*256*2)       // bf16 [I][H][D][S]
#define OFF_G    (OFF_V   + (size_t)ROWS*256*2)       // bf16 [I][H][S][D] (pre-sigmoided)
#define OFF_P    (OFF_G   + (size_t)ROWS*256*2)       // bf16 [65536][256] gated attn out

// ---------------- CDNA5 async global->LDS copy (ASYNCcnt-tracked) ----------------
// Generic pointers into LDS carry the LDS byte address in their low 32 bits
// (ISA 10.2), so a plain truncation gives VDST.
__device__ __forceinline__ void async_copy_b128(const bf16_t* __restrict__ gsrc,
                                                bf16_t* lds_dst) {
  unsigned loff = (unsigned)(size_t)lds_dst;
  asm volatile("global_load_async_to_lds_b128 %0, %1, off"
               :: "v"(loff), "v"(gsrc) : "memory");
}
__device__ __forceinline__ void wait_async0() {
  asm volatile("s_wait_asynccnt 0x0" ::: "memory");
}

// ---------------- WMMA fragment loaders (row-major A[M][K], Bt[N][K]) ----------------
__device__ __forceinline__ v16bf load_a_frag(const bf16_t* __restrict__ base, int ld) {
  int lane = threadIdx.x & 31;
  int half = lane >> 4, m = lane & 15;
  const bf16_t* p = base + (size_t)m * ld + 16 * half;
  v8bf lo = *(const v8bf*)p;          // K = 16*half + 0..7
  v8bf hi = *(const v8bf*)(p + 8);    // K = 16*half + 8..15
  v16bf r;
#pragma unroll
  for (int e = 0; e < 8; e++) { r[e] = lo[e]; r[e + 8] = hi[e]; }
  return r;
}

__device__ __forceinline__ v16bf load_b_frag(const bf16_t* __restrict__ base, int ld) {
  int lane = threadIdx.x & 31;
  int half = lane >> 4, n = lane & 15;
  return *(const v16bf*)(base + (size_t)n * ld + 16 * half);  // K = 16*half + 0..15
}

// ---------------- kernel 0: weight transpose + bf16 convert ----------------
__global__ __launch_bounds__(256) void prep_weights_kernel(
    const float* __restrict__ Wq, const float* __restrict__ Wk,
    const float* __restrict__ Wv, const float* __restrict__ Wg,
    const float* __restrict__ Wo,
    bf16_t* __restrict__ Wt, bf16_t* __restrict__ Wot) {
  int idx = blockIdx.x * 256 + threadIdx.x;
  if (idx < 1024 * 256) {
    int ng = idx >> 8, k = idx & 255;
    int proj = ng >> 8, nl = ng & 255;
    const float* W = (proj == 0) ? Wq : (proj == 1) ? Wk : (proj == 2) ? Wv : Wg;
    Wt[idx] = (bf16_t)W[k * 256 + nl];
  } else {
    int j = idx - 1024 * 256;
    int n = j >> 8, k = j & 255;
    Wot[j] = (bf16_t)Wo[k * 256 + n];
  }
}

// ---------------- kernel 1: LayerNorm (one wave per 256-channel row) ----------------
__global__ __launch_bounds__(256) void ln_kernel(
    const float* __restrict__ msa, const float* __restrict__ gamma,
    const float* __restrict__ beta, bf16_t* __restrict__ X) {
  int row  = blockIdx.x * 8 + (threadIdx.x >> 5);
  int lane = threadIdx.x & 31;
  const float* p = msa + (size_t)row * 256 + lane * 8;
  float4 a = *(const float4*)p;
  float4 b = *(const float4*)(p + 4);
  float x[8] = {a.x, a.y, a.z, a.w, b.x, b.y, b.z, b.w};
  float s = 0.f;
#pragma unroll
  for (int e = 0; e < 8; e++) s += x[e];
  for (int off = 1; off < 32; off <<= 1) s += __shfl_xor(s, off, 32);
  float mu = s * (1.f / 256.f);
  float vs = 0.f;
#pragma unroll
  for (int e = 0; e < 8; e++) { float d = x[e] - mu; vs += d * d; }
  for (int off = 1; off < 32; off <<= 1) vs += __shfl_xor(vs, off, 32);
  float inv = rsqrtf(vs * (1.f / 256.f) + 1e-5f);
  bf16_t* o = X + (size_t)row * 256 + lane * 8;
#pragma unroll
  for (int e = 0; e < 8; e++) {
    int c = lane * 8 + e;
    o[e] = (bf16_t)((x[e] - mu) * inv * gamma[c] + beta[c]);
  }
}

// ---------------- kernel 2: QKVG projection GEMM (M=65536, N=1024, K=256) --------
#define GM 128
#define GN 64
#define LDS_STRIDE 48   // 96 B rows: keeps 16/32-byte alignment of fragment loads

__global__ __launch_bounds__(256) void qkvg_gemm_kernel(
    const bf16_t* __restrict__ X,   // [65536][256]
    const bf16_t* __restrict__ Wt,  // [1024][256]  (B transposed: [N][K])
    const float*  __restrict__ bg,
    bf16_t* __restrict__ Qt, bf16_t* __restrict__ Kt,
    bf16_t* __restrict__ Vt, bf16_t* __restrict__ Gt) {
  __shared__ bf16_t As[GM][LDS_STRIDE];
  __shared__ bf16_t Bs[GN][LDS_STRIDE];
  const int tid = threadIdx.x;
  const int bm = blockIdx.x * GM;
  const int bn = blockIdx.y * GN;
  const int wid = tid >> 5;
  const int wm = wid >> 1, wn = wid & 1;
  v8f acc[2][2] = {};

  // A tile: 128 rows x 64 B = 512 x 16B chunks; B tile: 64 rows x 64 B = 256 chunks.
  const int rA0 = tid >> 2,           qA0 = tid & 3;
  const int rA1 = (tid + 256) >> 2,   qA1 = tid & 3;
  const int rB  = tid >> 2,           qB  = tid & 3;

  for (int k0 = 0; k0 < 256; k0 += 32) {
    __syncthreads();   // previous iteration's fragment reads complete
    async_copy_b128(&X [(size_t)(bm + rA0) * 256 + k0 + qA0 * 8], &As[rA0][qA0 * 8]);
    async_copy_b128(&X [(size_t)(bm + rA1) * 256 + k0 + qA1 * 8], &As[rA1][qA1 * 8]);
    async_copy_b128(&Wt[(size_t)(bn + rB ) * 256 + k0 + qB  * 8], &Bs[rB ][qB  * 8]);
    wait_async0();
    __syncthreads();
    v16bf afrag[2], bfrag[2];
#pragma unroll
    for (int i = 0; i < 2; i++) afrag[i] = load_a_frag(&As[wm * 32 + i * 16][0], LDS_STRIDE);
#pragma unroll
    for (int j = 0; j < 2; j++) bfrag[j] = load_b_frag(&Bs[wn * 32 + j * 16][0], LDS_STRIDE);
#pragma unroll
    for (int i = 0; i < 2; i++)
#pragma unroll
      for (int j = 0; j < 2; j++)
        acc[i][j] = WMMA_BF16(afrag[i], bfrag[j], acc[i][j]);
  }

  const int lane = tid & 31;
  const int half = lane >> 4, n = lane & 15;
  const int proj = bn >> 8;  // uniform per block (GN=64 divides 256)
#pragma unroll
  for (int i = 0; i < 2; i++)
#pragma unroll
    for (int j = 0; j < 2; j++)
#pragma unroll
      for (int v = 0; v < 8; v++) {
        int r = bm + wm * 32 + i * 16 + v + 8 * half;
        int c = bn + wn * 32 + j * 16 + n;
        int s = r >> 8, ii = r & 255;
        int rem = c & 255, h = rem >> 5, d = rem & 31;
        float val = acc[i][j][v];
        size_t qidx = (((size_t)ii * NHEAD + h) * S_DIM + s) * HDIM + d;
        if (proj == 0)      Qt[qidx] = (bf16_t)(val * QK_SCALE);
        else if (proj == 1) Kt[qidx] = (bf16_t)val;
        else if (proj == 2) Vt[(((size_t)ii * NHEAD + h) * HDIM + d) * S_DIM + s] = (bf16_t)val;
        else {
          float gv = val + bg[rem];
          Gt[qidx] = (bf16_t)(1.0f / (1.0f + __expf(-gv)));
        }
      }
}

// ---------------- kernel 3: fused column attention per (i,h) ----------------
// Block = 128 threads (4 waves); each wave owns 64 query rows (4 tiles of 16).
// K (16KB) and V^T (16KB) staged once per block via async copies; P strips 32KB.
__global__ __launch_bounds__(128) void attn_kernel(
    const bf16_t* __restrict__ Qt, const bf16_t* __restrict__ Kt,
    const bf16_t* __restrict__ Vt, const bf16_t* __restrict__ Gt,
    bf16_t* __restrict__ P_out) {
  __shared__ bf16_t Pb[4][16][256];   // per-wave P strip
  __shared__ bf16_t Kl[256][32];      // K tile  [t][d]
  __shared__ bf16_t Vl[32][256];      // V^T tile [d][t]
  const int ih = blockIdx.x;           // i*8 + h
  const bf16_t* Q  = Qt + (size_t)ih * S_DIM * HDIM;
  const bf16_t* K  = Kt + (size_t)ih * S_DIM * HDIM;
  const bf16_t* VT = Vt + (size_t)ih * HDIM * S_DIM;   // [d][t]
  const bf16_t* G  = Gt + (size_t)ih * S_DIM * HDIM;
  const int i_res = ih >> 3, h = ih & 7;
  const int tid = threadIdx.x;
  const int wid = tid >> 5, lane = tid & 31;
  const int half = lane >> 4, hn = lane & 15;

  // stage K: 256x32 = 1024 x 16B chunks;  V^T: 32x256 = 1024 chunks
#pragma unroll
  for (int c = 0; c < 8; c++) {
    int id = tid + c * 128;
    int kr = id >> 1, kq = id & 1;            // K row, 16B quarter (2 per row)
    async_copy_b128(&K[(size_t)kr * 32 + kq * 8], &Kl[kr][kq * 8]);
    int vd = id >> 5, vq = id & 31;           // V^T row d, 16B chunk (32 per row)
    async_copy_b128(&VT[(size_t)vd * 256 + vq * 8], &Vl[vd][vq * 8]);
  }
  wait_async0();
  __syncthreads();

  for (int rt = 0; rt < 4; rt++) {
    const int row0 = wid * 64 + rt * 16;
    // ---- pass 1: scores S = (Q*scale) K^T, 16x256 strip in 16 accumulators
    v16bf qa = load_a_frag(Q + (size_t)row0 * HDIM, HDIM);
    v8f sc[16];
#pragma unroll
    for (int nt = 0; nt < 16; nt++) {
      v16bf kb = load_b_frag(&Kl[nt * 16][0], 32);
      v8f z = {};
      sc[nt] = WMMA_BF16(qa, kb, z);
    }
    // ---- rowwise softmax stats (row = v + 8*half, cols across 16-lane group)
    float mx[8], sum[8];
#pragma unroll
    for (int v = 0; v < 8; v++) {
      float m = sc[0][v];
#pragma unroll
      for (int nt = 1; nt < 16; nt++) m = fmaxf(m, sc[nt][v]);
      for (int off = 1; off < 16; off <<= 1) m = fmaxf(m, __shfl_xor(m, off, 32));
      mx[v] = m;
    }
#pragma unroll
    for (int v = 0; v < 8; v++) {
      float s = 0.f;
#pragma unroll
      for (int nt = 0; nt < 16; nt++) {
        float p = __expf(sc[nt][v] - mx[v]);
        s += p;
        Pb[wid][v + 8 * half][nt * 16 + hn] = (bf16_t)p;
      }
      for (int off = 1; off < 16; off <<= 1) s += __shfl_xor(s, off, 32);
      sum[v] = s;
    }
    __syncthreads();   // order LDS writes before fragment reads (uniform trip count)
    // ---- pass 2: O = P V  (M=16, N=32, K=256)
    v8f o[2] = {};
#pragma unroll
    for (int kt = 0; kt < 8; kt++) {
      v16bf pa = load_a_frag(&Pb[wid][0][kt * 32], 256);
#pragma unroll
      for (int j = 0; j < 2; j++) {
        v16bf vb = load_b_frag(&Vl[j * 16][kt * 32], 256);
        o[j] = WMMA_BF16(pa, vb, o[j]);
      }
    }
    __syncthreads();
    // ---- epilogue: /rowsum, *sigmoid gate, store to [(s,i)][h*32+d]
#pragma unroll
    for (int j = 0; j < 2; j++)
#pragma unroll
      for (int v = 0; v < 8; v++) {
        int srow = row0 + v + 8 * half;
        int d = j * 16 + hn;
        float val = o[j][v] / sum[v];
        val *= (float)G[(size_t)srow * HDIM + d];
        P_out[((size_t)srow * I_DIM + i_res) * TOTAL + h * HDIM + d] = (bf16_t)val;
      }
  }
}

// ---------------- kernel 4: output GEMM + bias (M=65536, N=256, K=256) ----------
__global__ __launch_bounds__(256) void out_gemm_kernel(
    const bf16_t* __restrict__ A,    // [65536][256]
    const bf16_t* __restrict__ Wot,  // [256][256]  (B transposed)
    const float*  __restrict__ bo,
    float* __restrict__ out) {
  __shared__ bf16_t As[GM][LDS_STRIDE];
  __shared__ bf16_t Bs[GN][LDS_STRIDE];
  const int tid = threadIdx.x;
  const int bm = blockIdx.x * GM;
  const int bn = blockIdx.y * GN;
  const int wid = tid >> 5;
  const int wm = wid >> 1, wn = wid & 1;
  v8f acc[2][2] = {};

  const int rA0 = tid >> 2,         qA0 = tid & 3;
  const int rA1 = (tid + 256) >> 2, qA1 = tid & 3;
  const int rB  = tid >> 2,         qB  = tid & 3;

  for (int k0 = 0; k0 < 256; k0 += 32) {
    __syncthreads();
    async_copy_b128(&A  [(size_t)(bm + rA0) * 256 + k0 + qA0 * 8], &As[rA0][qA0 * 8]);
    async_copy_b128(&A  [(size_t)(bm + rA1) * 256 + k0 + qA1 * 8], &As[rA1][qA1 * 8]);
    async_copy_b128(&Wot[(size_t)(bn + rB ) * 256 + k0 + qB  * 8], &Bs[rB ][qB  * 8]);
    wait_async0();
    __syncthreads();
    v16bf afrag[2], bfrag[2];
#pragma unroll
    for (int i = 0; i < 2; i++) afrag[i] = load_a_frag(&As[wm * 32 + i * 16][0], LDS_STRIDE);
#pragma unroll
    for (int j = 0; j < 2; j++) bfrag[j] = load_b_frag(&Bs[wn * 32 + j * 16][0], LDS_STRIDE);
#pragma unroll
    for (int i = 0; i < 2; i++)
#pragma unroll
      for (int j = 0; j < 2; j++)
        acc[i][j] = WMMA_BF16(afrag[i], bfrag[j], acc[i][j]);
  }

  const int lane = tid & 31;
  const int half = lane >> 4, n = lane & 15;
#pragma unroll
  for (int i = 0; i < 2; i++)
#pragma unroll
    for (int j = 0; j < 2; j++)
#pragma unroll
      for (int v = 0; v < 8; v++) {
        int r = bm + wm * 32 + i * 16 + v + 8 * half;
        int c = bn + wn * 32 + j * 16 + n;
        out[(size_t)r * 256 + c] = acc[i][j][v] + bo[c];
      }
}

// ---------------- launcher ----------------
extern "C" void kernel_launch(void* const* d_in, const int* in_sizes, int n_in,
                              void* d_out, int out_size, void* d_ws, size_t ws_size,
                              hipStream_t stream) {
  const float* msa      = (const float*)d_in[0];
  const float* ln_gamma = (const float*)d_in[1];
  const float* ln_beta  = (const float*)d_in[2];
  const float* Wq       = (const float*)d_in[3];
  const float* Wk       = (const float*)d_in[4];
  const float* Wv       = (const float*)d_in[5];
  const float* Wg       = (const float*)d_in[6];
  const float* bg       = (const float*)d_in[7];
  const float* Wo       = (const float*)d_in[8];
  const float* bo       = (const float*)d_in[9];
  float* out = (float*)d_out;

  char* ws = (char*)d_ws;
  bf16_t* X   = (bf16_t*)(ws + OFF_X);
  bf16_t* Wt  = (bf16_t*)(ws + OFF_WT);
  bf16_t* Wot = (bf16_t*)(ws + OFF_WOT);
  bf16_t* Qt  = (bf16_t*)(ws + OFF_Q);
  bf16_t* Kt  = (bf16_t*)(ws + OFF_K);
  bf16_t* Vt  = (bf16_t*)(ws + OFF_V);
  bf16_t* Gt  = (bf16_t*)(ws + OFF_G);
  bf16_t* Pg  = (bf16_t*)(ws + OFF_P);

  prep_weights_kernel<<<(1024 * 256 + 256 * 256) / 256, 256, 0, stream>>>(
      Wq, Wk, Wv, Wg, Wo, Wt, Wot);
  ln_kernel<<<ROWS / 8, 256, 0, stream>>>(msa, ln_gamma, ln_beta, X);
  qkvg_gemm_kernel<<<dim3(ROWS / GM, 1024 / GN), 256, 0, stream>>>(
      X, Wt, bg, Qt, Kt, Vt, Gt);
  attn_kernel<<<I_DIM * NHEAD, 128, 0, stream>>>(Qt, Kt, Vt, Gt, Pg);
  out_gemm_kernel<<<dim3(ROWS / GM, TOTAL / GN), 256, 0, stream>>>(
      Pg, Wot, bo, out);
}
